// MPNN_18485539242873
// MI455X (gfx1250) — matrix operand order repaired and checked by reading, not compiled, not run on hardware
//
#include <hip/hip_runtime.h>
#include <hip/hip_bf16.h>

// ---------------------------------------------------------------------------
// MPNN forward for MI455X (gfx1250, wave32, WMMA).
// Edge MLP (enc 32x32, m1 96x32, m2 32x32) runs on v_wmma_f32_16x16x32_f16,
// 16 edges per wave per tile. HBM-bound problem (~0.5 GB moved, ~17 GFLOP),
// so the goal is: stream h_msg once, keep h_node gather table + aggr atomic
// target in L2 (3.2 MB f16 / 6.4 MB f32), do all matmul work on tensor cores.
// ---------------------------------------------------------------------------

typedef _Float16 v16h __attribute__((ext_vector_type(16)));
typedef _Float16 v8h  __attribute__((ext_vector_type(8)));
typedef float    v8f  __attribute__((ext_vector_type(8)));

#define H 32
#define FPAD 104   // padded f16 row stride of LDS feature buffer (16B-multiple)

static __device__ __forceinline__ float lrelu(float x) {
    return x > 0.0f ? x : 0.01f * x;
}

// Load a 16x32 f16 A-operand tile from the per-wave LDS feature buffer.
// A-layout (16-bit 16x32): lane L<16 holds row L, K={kb..kb+7, kb+16..kb+23};
// lane L>=16 holds row L-16, K={kb+8..kb+15, kb+24..kb+31}.
static __device__ __forceinline__ v16h lds_loadA(const _Float16* F, int r, int hw, int kbase) {
    const int o = r * FPAD + kbase + hw * 8;
    v8h lo = *reinterpret_cast<const v8h*>(F + o);
    v8h hi = *reinterpret_cast<const v8h*>(F + o + 16);
    v16h a;
#pragma unroll
    for (int i = 0; i < 8; ++i) { a[i] = lo[i]; a[i + 8] = hi[i]; }
    return a;
}

static __device__ __forceinline__ v8f wmma32(v16h a, v16h b, v8f c) {
    return __builtin_amdgcn_wmma_f32_16x16x32_f16(false, a, false, b, (short)0, c, false, false);
}

// ---------------------------------------------------------------------------
// Kernel 0: swizzle weights (f32 row-major [K][N]) into WMMA B-operand order.
// B-layout per (ktile,ntile) set: lane L<16 -> col N=L, K=kt*32+{0..7,16..23} packed
// pairs in 8 VGPRs; lane L>=16 -> col N=L-16, K=kt*32+{8..15,24..31}.
// Sets: 0..1 = W_enc[2:34] (nt 0/1); 2..7 = W_m1 (kt 0..2 x nt 0/1);
//       8..9 = W_m2 (nt 0/1).  512 f16 per set.
// ---------------------------------------------------------------------------
__global__ void mpnn_prep_weights(const float* __restrict__ Wenc,
                                  const float* __restrict__ Wm1,
                                  const float* __restrict__ Wm2,
                                  _Float16* __restrict__ wsB) {
    for (int t = threadIdx.x; t < 10 * 512; t += blockDim.x) {
        const int s   = t >> 9;
        const int rem = t & 511;
        const int L   = rem >> 4;      // lane
        const int tt  = (rem & 15) + (rem & 8);      // k offset: 0..7, 16..23
        const int hiK = (L >= 16) ? 8 : 0;
        const int nl  = L & 15;
        float v;
        if (s < 2) {                       // W_enc rows 2..33 (h_msg part)
            const int n = s * 16 + nl;
            v = Wenc[(2 + hiK + tt) * H + n];
        } else if (s < 8) {                // W_m1: 96x32
            const int q = s - 2, kt = q >> 1, nt = q & 1;
            v = Wm1[(kt * 32 + hiK + tt) * H + nt * 16 + nl];
        } else {                           // W_m2: 32x32
            const int nt = s - 8;
            v = Wm2[(hiK + tt) * H + nt * 16 + nl];
        }
        wsB[t] = (_Float16)v;
    }
}

// ---------------------------------------------------------------------------
// Kernel 1: h_node = x @ W_in + b_in (f32 + f16 copies), zero aggr.
// ---------------------------------------------------------------------------
__global__ void mpnn_node_embed(const float* __restrict__ x,
                                const float* __restrict__ Win,
                                const float* __restrict__ bin,
                                float* __restrict__ hn32,
                                _Float16* __restrict__ hn16,
                                float* __restrict__ aggr, int N) {
    const int total = N * H;
    for (int t = blockIdx.x * blockDim.x + threadIdx.x; t < total;
         t += gridDim.x * blockDim.x) {
        const int nd = t / H, k = t % H;
        const float h = x[nd * 2 + 0] * Win[k] + x[nd * 2 + 1] * Win[H + k] + bin[k];
        hn32[t] = h;
        hn16[t] = (_Float16)h;
        aggr[t] = 0.0f;
    }
}

// ---------------------------------------------------------------------------
// Kernel 2: the edge MLP on WMMA. One wave handles 16 edges per tile.
// ---------------------------------------------------------------------------
__global__ __launch_bounds__(256) void mpnn_edge_kernel(
    const int* __restrict__ ei,          // [2][E], row0=src, row1=dst
    const float* __restrict__ hm,        // h_msg [E][32]
    const float* __restrict__ ym,        // y_msg [E][2]
    const float* __restrict__ Wenc, const float* __restrict__ benc,
    const float* __restrict__ bm1, const float* __restrict__ bm2,
    const float* __restrict__ Wdec, const float* __restrict__ bdec,
    const _Float16* __restrict__ wsB,
    const _Float16* __restrict__ hn16,
    float* __restrict__ aggr,
    float* __restrict__ out_m,           // [E][32]
    float* __restrict__ out_y,           // [E][2]
    int E, int nTiles) {
    __shared__ _Float16 ldsF[8][16 * FPAD];  // per-wave [16 edges][h_i|h_j|enc]
    __shared__ int ldsIdx[8][16];            // dst index per edge row

    const int tid = threadIdx.x;
    const int w = tid >> 5;          // wave in block
    const int L = tid & 31;          // lane
    const int r = L & 15;            // row / col-in-ntile
    const int hw = L >> 4;           // half-wave

    _Float16* F = &ldsF[w][0];
    int* idxW = &ldsIdx[w][0];

    // ---- per-wave constants -------------------------------------------------
    v16h Bv[10];
#pragma unroll
    for (int s = 0; s < 10; ++s) {
        const v8h* p = reinterpret_cast<const v8h*>(wsB + s * 512 + L * 16);
        v8h lo = p[0], hi = p[1];
#pragma unroll
        for (int i = 0; i < 8; ++i) { Bv[s][i] = lo[i]; Bv[s][i + 8] = hi[i]; }
    }
    // C-init constants for this lane's two output columns (n = r, n = 16+r)
    const float we0_0 = Wenc[r],          we0_1 = Wenc[16 + r];
    const float we1_0 = Wenc[H + r],      we1_1 = Wenc[H + 16 + r];
    const float be_0  = benc[r],          be_1  = benc[16 + r];
    const float b1_0  = bm1[r],           b1_1  = bm1[16 + r];
    const float b2_0  = bm2[r],           b2_1  = bm2[16 + r];
    const float wd00 = Wdec[r * 2], wd01 = Wdec[r * 2 + 1];
    const float wd10 = Wdec[(16 + r) * 2], wd11 = Wdec[(16 + r) * 2 + 1];
    const float bd0 = bdec[0], bd1 = bdec[1];

    const int waveGlobal = blockIdx.x * 8 + w;
    const int waveCount = gridDim.x * 8;

    for (int tile = waveGlobal; tile < nTiles; tile += waveCount) {
        const int e0 = tile * 16;
        __builtin_amdgcn_wave_barrier();

        // ---- gather h_i (lanes 0-15, dst) / h_j (lanes 16-31, src) ---------
        const int idx = (hw == 0) ? ei[E + e0 + r] : ei[e0 + r];
        if (hw == 0) idxW[r] = idx;
        {
            const v8h* hp = reinterpret_cast<const v8h*>(hn16 + (size_t)idx * H);
            v8h* fr = reinterpret_cast<v8h*>(F + r * FPAD + hw * 32);
            fr[0] = hp[0]; fr[1] = hp[1]; fr[2] = hp[2]; fr[3] = hp[3];
        }

        // ---- enc = h_msg @ W_enc[2:34]  (+ y-part & bias folded into C) ----
        const float* mp = hm + (size_t)(e0 + r) * H + hw * 8;
        v16h aE;
#pragma unroll
        for (int i = 0; i < 8; ++i) aE[i] = (_Float16)mp[i];
#pragma unroll
        for (int i = 0; i < 8; ++i) aE[i + 8] = (_Float16)mp[16 + i];

        v8f cE0, cE1;
#pragma unroll
        for (int v = 0; v < 8; ++v) {
            const float2 yy = *reinterpret_cast<const float2*>(ym + (size_t)(e0 + v + hw * 8) * 2);
            cE0[v] = yy.x * we0_0 + yy.y * we1_0 + be_0;
            cE1[v] = yy.x * we0_1 + yy.y * we1_1 + be_1;
        }
        cE0 = wmma32(aE, Bv[0], cE0);
        cE1 = wmma32(aE, Bv[1], cE1);
#pragma unroll
        for (int v = 0; v < 8; ++v) {
            F[(v + hw * 8) * FPAD + 64 + r]      = (_Float16)cE0[v];
            F[(v + hw * 8) * FPAD + 64 + 16 + r] = (_Float16)cE1[v];
        }
        __builtin_amdgcn_wave_barrier();   // LDS in-order within wave; fence compiler

        // ---- m1 = lrelu([h_i|h_j|enc] @ W_m1 + b_m1), K = 96 ---------------
        v8f c10, c11;
#pragma unroll
        for (int v = 0; v < 8; ++v) { c10[v] = b1_0; c11[v] = b1_1; }
#pragma unroll
        for (int kt = 0; kt < 3; ++kt) {
            v16h a = lds_loadA(F, r, hw, kt * 32);
            c10 = wmma32(a, Bv[2 + kt * 2 + 0], c10);
            c11 = wmma32(a, Bv[2 + kt * 2 + 1], c11);
        }
#pragma unroll
        for (int v = 0; v < 8; ++v) {   // activation -> f16, reuse cols 0..31
            F[(v + hw * 8) * FPAD + r]      = (_Float16)lrelu(c10[v]);
            F[(v + hw * 8) * FPAD + 16 + r] = (_Float16)lrelu(c11[v]);
        }
        __builtin_amdgcn_wave_barrier();

        // ---- m = lrelu(m1 @ W_m2 + b_m2), K = 32 ---------------------------
        v16h a2 = lds_loadA(F, r, hw, 0);
        v8f c20, c21;
#pragma unroll
        for (int v = 0; v < 8; ++v) { c20[v] = b2_0; c21[v] = b2_1; }
        c20 = wmma32(a2, Bv[8], c20);
        c21 = wmma32(a2, Bv[9], c21);
        __builtin_amdgcn_wave_barrier();

        // ---- outputs: m, aggr atomics, decoder softmax ---------------------
#pragma unroll
        for (int v = 0; v < 8; ++v) {
            const int M = v + hw * 8;
            const size_t e = (size_t)e0 + M;
            const int di = idxW[M];
            const float m0 = lrelu(c20[v]);
            const float m1v = lrelu(c21[v]);
            out_m[e * H + r]      = m0;
            out_m[e * H + 16 + r] = m1v;
            unsafeAtomicAdd(&aggr[(size_t)di * H + r],      m0);
            unsafeAtomicAdd(&aggr[(size_t)di * H + 16 + r], m1v);
            // decoder logits: half-wave reduction over 16 lanes (2 cols/lane)
            float p0 = m0 * wd00 + m1v * wd10;
            float p1 = m0 * wd01 + m1v * wd11;
            p0 += __shfl_xor(p0, 1, 16); p0 += __shfl_xor(p0, 2, 16);
            p0 += __shfl_xor(p0, 4, 16); p0 += __shfl_xor(p0, 8, 16);
            p1 += __shfl_xor(p1, 1, 16); p1 += __shfl_xor(p1, 2, 16);
            p1 += __shfl_xor(p1, 4, 16); p1 += __shfl_xor(p1, 8, 16);
            if (r == 0) {
                const float l0 = p0 + bd0, l1 = p1 + bd1;
                const float mx = fmaxf(l0, l1);
                const float z0 = __expf(l0 - mx), z1 = __expf(l1 - mx);
                const float inv = 1.0f / (z0 + z1);
                *reinterpret_cast<float2*>(out_y + e * 2) = make_float2(z0 * inv, z1 * inv);
            }
        }
    }
}

// ---------------------------------------------------------------------------
// Kernel 3: node update + beliefs (tiny: 0.2 GFLOP, plain VALU f32).
// ---------------------------------------------------------------------------
__global__ __launch_bounds__(256) void mpnn_node_update(
    const float* __restrict__ hn, const float* __restrict__ aggr,
    const float* __restrict__ Wu, const float* __restrict__ bu,
    const float* __restrict__ Wb, const float* __restrict__ bb,
    float* __restrict__ outb, int N) {
    __shared__ float sWu[64 * H];
    __shared__ float sbu[H];
    __shared__ float sWb[H * 2];
    __shared__ float sbb2[2];
    for (int t = threadIdx.x; t < 64 * H; t += blockDim.x) sWu[t] = Wu[t];
    if (threadIdx.x < H)     sbu[threadIdx.x]  = bu[threadIdx.x];
    if (threadIdx.x < H * 2) sWb[threadIdx.x]  = Wb[threadIdx.x];
    if (threadIdx.x < 2)     sbb2[threadIdx.x] = bb[threadIdx.x];
    __syncthreads();

    for (int nd = blockIdx.x * blockDim.x + threadIdx.x; nd < N;
         nd += gridDim.x * blockDim.x) {
        float acc[H];
#pragma unroll
        for (int j = 0; j < H; ++j) acc[j] = sbu[j];
        const float* hp = hn + (size_t)nd * H;
        const float* ap = aggr + (size_t)nd * H;
        for (int k = 0; k < H; ++k) {
            const float hv = hp[k];
#pragma unroll
            for (int j = 0; j < H; ++j) acc[j] += hv * sWu[k * H + j];
        }
        for (int k = 0; k < H; ++k) {
            const float av = ap[k];
#pragma unroll
            for (int j = 0; j < H; ++j) acc[j] += av * sWu[(H + k) * H + j];
        }
        float l0 = sbb2[0], l1 = sbb2[1];
#pragma unroll
        for (int j = 0; j < H; ++j) {
            const float h = lrelu(acc[j]);
            l0 += h * sWb[j * 2];
            l1 += h * sWb[j * 2 + 1];
        }
        const float mx = fmaxf(l0, l1);
        const float z0 = __expf(l0 - mx), z1 = __expf(l1 - mx);
        const float inv = 1.0f / (z0 + z1);
        outb[(size_t)nd * 2]     = z0 * inv;
        outb[(size_t)nd * 2 + 1] = z1 * inv;
    }
}

// ---------------------------------------------------------------------------
static inline size_t align256(size_t x) { return (x + 255) & ~(size_t)255; }

extern "C" void kernel_launch(void* const* d_in, const int* in_sizes, int n_in,
                              void* d_out, int out_size, void* d_ws, size_t ws_size,
                              hipStream_t stream) {
    const float* x    = (const float*)d_in[0];
    const int*   ei   = (const int*)d_in[1];
    const float* hmsg = (const float*)d_in[2];
    const float* ymsg = (const float*)d_in[3];
    const float* Win  = (const float*)d_in[4];
    const float* bin  = (const float*)d_in[5];
    const float* Wenc = (const float*)d_in[6];
    const float* benc = (const float*)d_in[7];
    const float* Wm1  = (const float*)d_in[8];
    const float* bm1  = (const float*)d_in[9];
    const float* Wm2  = (const float*)d_in[10];
    const float* bm2  = (const float*)d_in[11];
    const float* Wu   = (const float*)d_in[12];
    const float* bu   = (const float*)d_in[13];
    const float* Wdec = (const float*)d_in[14];
    const float* bdec = (const float*)d_in[15];
    const float* Wbel = (const float*)d_in[16];
    const float* bbel = (const float*)d_in[17];

    const int N = in_sizes[0] / 2;       // 50000
    const int E = in_sizes[2] / H;       // 1600000

    // workspace layout
    char* ws = (char*)d_ws;
    size_t off = 0;
    float*    hn32 = (float*)(ws + off);    off += align256((size_t)N * H * sizeof(float));
    _Float16* hn16 = (_Float16*)(ws + off); off += align256((size_t)N * H * sizeof(_Float16));
    float*    aggr = (float*)(ws + off);    off += align256((size_t)N * H * sizeof(float));
    _Float16* wsB  = (_Float16*)(ws + off); off += align256((size_t)10 * 512 * sizeof(_Float16));
    (void)ws_size; (void)n_in; (void)out_size;

    float* out_m = (float*)d_out;
    float* out_y = out_m + (size_t)E * H;
    float* out_b = out_y + (size_t)E * 2;

    mpnn_prep_weights<<<1, 256, 0, stream>>>(Wenc, Wm1, Wm2, wsB);
    {
        const int total = N * H;
        const int blocks = (total + 255) / 256;
        mpnn_node_embed<<<blocks, 256, 0, stream>>>(x, Win, bin, hn32, hn16, aggr, N);
    }
    {
        const int nTiles = E / 16;
        mpnn_edge_kernel<<<2048, 256, 0, stream>>>(ei, hmsg, ymsg,
                                                   Wenc, benc, bm1, bm2, Wdec, bdec,
                                                   wsB, hn16, aggr, out_m, out_y,
                                                   E, nTiles);
    }
    mpnn_node_update<<<256, 256, 0, stream>>>(hn32, aggr, Wu, bu, Wbel, bbel, out_b, N);
}
